// PocatModel_31756988187265
// MI455X (gfx1250) — matrix operand order repaired and checked by reading, not compiled, not run on hardware
//
#include <hip/hip_runtime.h>
#include <hip/hip_bf16.h>

// =====================================================================
// PocatModel forward for gfx1250 (MI455X), wave32 + WMMA f32_16x16x32_f16.
//
// Input flattening assumption (setup_inputs() dict order, depth-first):
//   0 nodes[8,512,16]  1 scalar[8,4]  2 matrix[8,512,512]
//   3 emb_w[16,256]    4 emb_b[256]
//   5+10*i  sp layer i: wq,wk,wv,wo,bo,n1,n2,l1,l2,l3
//   35+10*i gl layer i: same
//   65+2*i  sf i: w,b        71+2*i gf i: w,b
//   75.. prompt: sw1,sb1,sw2,sb2,mw1,mb1,mw2,mb2,fw,fb,lnw,lnb
// Output: sp [8,512,256] float32.  Workspace: ~74 MB.
//
// WMMA data path: A fragments are loaded DIRECTLY from global memory
// (each lane's 16 halves come from two contiguous 8-float runs of its own
// output row -> global_load_b128 + cvt).  B tiles are staged in LDS in
// fragment-linear order: one 32-byte v16h record per (subtile,lane), so a
// fragment load is two ds_load_b128 instead of 16 ds_load_u16.
// =====================================================================

typedef __attribute__((ext_vector_type(16))) _Float16 v16h;
typedef __attribute__((ext_vector_type(8)))  float    v8f;

#define E_  256
#define H_  8
#define D_  32
#define NSP 512
#define NGL 513
#define NPAD 528
#define INNER 768
#define BATCH 8
#define XS   ((long)NPAD * E_)       // per-batch activation stride (floats)
#define FFS  ((long)NPAD * INNER)    // per-batch ffn stride
#define LDS_S 576                    // score row stride (9 tiles of 64)
#define SH   ((long)LDS_S * LDS_S)   // per-head score stride

static inline int cdiv(int a, int b) { return (a + b - 1) / b; }

// A-fragment k-offsets (contiguous order, same as B, lane h = lane>>4):
//   j in [0,8)  -> k = 16h + j        (contiguous run)
//   j in [8,16) -> k = 16h + j (contiguous run)
// B-fragment (32x16): element j -> k = j + 16h, n = n0 + (lane&15).

__device__ __forceinline__ v8f wmma_step(v16h a, v16h b, v8f c) {
  return __builtin_amdgcn_wmma_f32_16x16x32_f16(false, a, false, b,
                                                (short)0, c, false, false);
}

// Load A fragment directly from global: row `rowp` (guard row<M outside),
// k-chunk base k0, fast when k0+32<=K.
__device__ __forceinline__ v16h a_frag_global(const float* __restrict__ rowp,
                                              bool rowOk, int k0, int K, int h)
{
  v16h a;
  if (rowOk && (k0 + 32 <= K)) {
    const float* p = rowp + k0 + 16 * h;
#pragma unroll
    for (int j = 0; j < 8; ++j) a[j] = (_Float16)p[j];
#pragma unroll
    for (int j = 0; j < 8; ++j) a[8 + j] = (_Float16)p[8 + j];
  } else {
#pragma unroll
    for (int j = 0; j < 16; ++j) {
      int k = k0 + 16 * h + j;
      a[j] = (_Float16)((rowOk && k < K) ? rowp[k] : 0.f);
    }
  }
  return a;
}

// ---------------- generic GEMM: C = A@W (+bias)(+R)(relu) ----------------
// 128 threads (4 waves), 64x64 tile.  kChunk>0 => split-K w/ atomicAdd.
__global__ __launch_bounds__(128)
void wmma_gemm(const float* __restrict__ A, long sAb, int lda,
               const float* __restrict__ W, int ldb,
               float* __restrict__ C, long sCb, int ldc,
               const float* __restrict__ bias,
               const float* __restrict__ R, long sRb, int ldr,
               int M, int N, int K, int act, int kChunk)
{
  __shared__ __align__(32) _Float16 Bs[2048];  // 4 subtiles x 32 lanes x 16 halves
  const int tid = threadIdx.x, wave = tid >> 5, lane = tid & 31;
  const int mloc = lane & 15, h = lane >> 4;

  int batch, k0beg, k0end;
  if (kChunk > 0) { batch = 0; k0beg = blockIdx.z * kChunk;
                    k0end = (K < k0beg + kChunk) ? K : (k0beg + kChunk); }
  else            { batch = blockIdx.z; k0beg = 0; k0end = K; }

  const float* Ab = A + (long)batch * sAb;
  float*       Cb = C + (long)batch * sCb;
  const float* Rb = R ? (R + (long)batch * sRb) : nullptr;

  const int tileM = blockIdx.y * 64;
  const int tileN = blockIdx.x * 64;

  const int gmA = tileM + wave * 16 + mloc;          // A row owned by this lane
  const bool aOk = (gmA < M);
  const float* Arow = Ab + (long)gmA * lda;
  const int gnB = tileN + wave * 16 + mloc;          // B column staged (t = wave)
  const bool bOk = (gnB < N);

  v8f acc[4] = {};

  for (int k0 = k0beg; k0 < k0end; k0 += 32) {
    if (k0 + 32 < k0end)
      __builtin_prefetch(&W[(long)(k0 + 32) * ldb + tileN], 0, 3);
    // stage this thread's B record: element j -> W[k0 + j + 16h][gnB]
    v16h bv;
    if (bOk && (k0 + 32 <= K)) {
      const float* p = W + (long)(k0 + 16 * h) * ldb + gnB;
#pragma unroll
      for (int j = 0; j < 16; ++j) bv[j] = (_Float16)p[(long)j * ldb];
    } else {
#pragma unroll
      for (int j = 0; j < 16; ++j) {
        int gk = k0 + j + 16 * h;
        bv[j] = (_Float16)((bOk && gk < K) ? W[(long)gk * ldb + gnB] : 0.f);
      }
    }
    *(v16h*)&Bs[tid << 4] = bv;                      // 2x ds_store_b128
    v16h av = a_frag_global(Arow, aOk, k0, K, h);    // global -> regs, no LDS
    __syncthreads();
#pragma unroll
    for (int t = 0; t < 4; ++t) {
      v16h bf = *(const v16h*)&Bs[(t * 32 + lane) << 4];  // 2x ds_load_b128
      acc[t] = wmma_step(av, bf, acc[t]);
    }
    __syncthreads();
  }

  const int nlo = lane & 15;
#pragma unroll
  for (int t = 0; t < 4; ++t)
#pragma unroll
    for (int r = 0; r < 8; ++r) {
      int gm = tileM + wave * 16 + r + 8 * h;
      int gn = tileN + t * 16 + nlo;
      if (gm < M && gn < N) {
        float v = acc[t][r];
        if (kChunk > 0) { atomicAdd(&Cb[(long)gm * ldc + gn], v); }
        else {
          if (bias) v += bias[gn];
          if (Rb)   v += Rb[(long)gm * ldr + gn];
          if (act == 1) v = fmaxf(v, 0.f);
          Cb[(long)gm * ldc + gn] = v;
        }
      }
    }
}

// ---------- attention scores: S = (Q K^T)*scale (+mask), K-dim = 32 ------
__global__ __launch_bounds__(128)
void wmma_attn_scores(const float* __restrict__ Q, const float* __restrict__ Kmat,
                      float* __restrict__ S, const unsigned char* __restrict__ mask,
                      int n, int ldx, int ldS, int mstride, float scale)
{
  const int hh = blockIdx.z;
  const float* Qh = Q + hh * D_;
  const float* Kh = Kmat + hh * D_;
  float* Sh = S + (long)hh * SH;
  __shared__ __align__(32) _Float16 Bs[2048];
  const int tid = threadIdx.x, wave = tid >> 5, lane = tid & 31;
  const int mloc = lane & 15, h = lane >> 4;
  const int tileM = blockIdx.y * 64, tileN = blockIdx.x * 64;

  // stage B record: K-matrix row gnB, head-dims 16h..16h+15 (contiguous!)
  const int gnB = tileN + wave * 16 + mloc;
  v16h bv;
  if (gnB < n) {
    const float* p = Kh + (long)gnB * ldx + 16 * h;
#pragma unroll
    for (int j = 0; j < 16; ++j) bv[j] = (_Float16)p[j];
  } else {
#pragma unroll
    for (int j = 0; j < 16; ++j) bv[j] = (_Float16)0.f;
  }
  *(v16h*)&Bs[tid << 4] = bv;

  const int gmA = tileM + wave * 16 + mloc;
  v16h av = a_frag_global(Qh + (long)gmA * ldx, gmA < n, 0, D_, h);
  __syncthreads();

  v8f acc[4] = {};
#pragma unroll
  for (int t = 0; t < 4; ++t) {
    v16h bf = *(const v16h*)&Bs[(t * 32 + lane) << 4];
    acc[t] = wmma_step(av, bf, acc[t]);
  }

  const int nlo = lane & 15;
#pragma unroll
  for (int t = 0; t < 4; ++t)
#pragma unroll
    for (int r = 0; r < 8; ++r) {
      int gm = tileM + wave * 16 + r + 8 * h;
      int gn = tileN + t * 16 + nlo;
      if (gm < n && gn < n) {
        float v = acc[t][r] * scale;
        if (mask && !mask[(long)gm * mstride + gn]) v = -1e9f;
        Sh[(long)gm * ldS + gn] = v;
      }
    }
}

// ---------- attention context: O = S @ V (N = 32 per head) --------------
__global__ __launch_bounds__(128)
void wmma_attn_ctx(const float* __restrict__ S, const float* __restrict__ V,
                   float* __restrict__ O, int n, int ldS, int ldx)
{
  const int hh = blockIdx.z;
  const float* Sh = S + (long)hh * SH;
  const float* Vh = V + hh * D_;
  float* Oh = O + hh * D_;
  __shared__ __align__(32) _Float16 Bs[1024];   // 2 subtiles x 32 lanes x 16
  const int tid = threadIdx.x, wave = tid >> 5, lane = tid & 31;
  const int mloc = lane & 15, h = lane >> 4;
  const int tileM = blockIdx.y * 64;

  const int gmA = tileM + wave * 16 + mloc;
  const bool aOk = (gmA < n);
  const float* Srow = Sh + (long)gmA * ldS;

  v8f acc[2] = {};
  for (int k0 = 0; k0 < n; k0 += 32) {
    if (wave < 2) {                              // stage V records (t = wave)
      const int gnB = wave * 16 + mloc;          // V column 0..31
      v16h bv;
#pragma unroll
      for (int j = 0; j < 16; ++j) {
        int gk = k0 + j + 16 * h;
        bv[j] = (_Float16)((gk < n) ? Vh[(long)gk * ldx + gnB] : 0.f);
      }
      *(v16h*)&Bs[tid << 4] = bv;
    }
    v16h av = a_frag_global(Srow, aOk, k0, n, h);
    __syncthreads();
#pragma unroll
    for (int t = 0; t < 2; ++t) {
      v16h bf = *(const v16h*)&Bs[(t * 32 + lane) << 4];
      acc[t] = wmma_step(av, bf, acc[t]);
    }
    __syncthreads();
  }
  const int nlo = lane & 15;
#pragma unroll
  for (int t = 0; t < 2; ++t)
#pragma unroll
    for (int r = 0; r < 8; ++r) {
      int gm = tileM + wave * 16 + r + 8 * h;
      int gn = t * 16 + nlo;
      if (gm < n) Oh[(long)gm * ldx + gn] = acc[t][r];
    }
}

// ---------- exact top-k (k=256 of 512) + softmax, per score row ----------
__global__ __launch_bounds__(256)
void topk_softmax_512(float* __restrict__ S, int ldS)
{
  float* row = S + (long)blockIdx.z * SH + (long)blockIdx.x * ldS;
  __shared__ float v[512];
  __shared__ float red[256];
  const int t = threadIdx.x;
  v[t] = row[t]; v[t + 256] = row[t + 256];
  __syncthreads();
  for (int k = 2; k <= 512; k <<= 1)          // bitonic sort, descending
    for (int j = k >> 1; j > 0; j >>= 1) {
      for (int base = t; base < 512; base += 256) {
        int ixj = base ^ j;
        if (ixj > base) {
          bool desc = ((base & k) == 0);
          float a = v[base], b = v[ixj];
          if (desc ? (a < b) : (a > b)) { v[base] = b; v[ixj] = a; }
        }
      }
      __syncthreads();
    }
  const float thresh = v[255];                // 256th largest
  const float vmax = v[0];
  float x0 = row[t], x1 = row[t + 256];
  float e0 = (x0 >= thresh) ? __expf(x0 - vmax) : 0.f;
  float e1 = (x1 >= thresh) ? __expf(x1 - vmax) : 0.f;
  red[t] = e0 + e1; __syncthreads();
  for (int s = 128; s > 0; s >>= 1) { if (t < s) red[t] += red[t + s]; __syncthreads(); }
  const float inv = 1.f / red[0];
  row[t] = e0 * inv; row[t + 256] = e1 * inv;
}

// ---------- plain (masked-already) softmax over row of length n ----------
__global__ __launch_bounds__(256)
void softmax_row(float* __restrict__ S, int n, int ldS)
{
  float* row = S + (long)blockIdx.z * SH + (long)blockIdx.x * ldS;
  __shared__ float red[256];
  const int t = threadIdx.x;
  float m = -3.4e38f;
  for (int i = t; i < n; i += 256) m = fmaxf(m, row[i]);
  red[t] = m; __syncthreads();
  for (int s = 128; s > 0; s >>= 1) { if (t < s) red[t] = fmaxf(red[t], red[t + s]); __syncthreads(); }
  m = red[0]; __syncthreads();
  float sum = 0.f;
  for (int i = t; i < n; i += 256) sum += __expf(row[i] - m);
  red[t] = sum; __syncthreads();
  for (int s = 128; s > 0; s >>= 1) { if (t < s) red[t] += red[t + s]; __syncthreads(); }
  const float inv = 1.f / red[0];
  for (int i = t; i < n; i += 256) row[i] = __expf(row[i] - m) * inv;
}

// ---------- RMS norm over E=256 cols ----------
__global__ __launch_bounds__(256)
void rms_rows(const float* __restrict__ X, float* __restrict__ Y,
              const float* __restrict__ w, long bstride)
{
  const float* x = X + (long)blockIdx.z * bstride + (long)blockIdx.x * E_;
  float* y = Y + (long)blockIdx.z * bstride + (long)blockIdx.x * E_;
  __shared__ float red[256];
  const int t = threadIdx.x;
  float v = x[t];
  red[t] = v * v; __syncthreads();
  for (int s = 128; s > 0; s >>= 1) { if (t < s) red[t] += red[t + s]; __syncthreads(); }
  y[t] = v * rsqrtf(red[0] / (float)E_ + 1e-6f) * w[t];
}

// ---------- LayerNorm + ReLU (prompt head) ----------
__global__ __launch_bounds__(256)
void ln_relu_rows(const float* __restrict__ X, float* __restrict__ Y,
                  const float* __restrict__ w, const float* __restrict__ bb)
{
  const float* x = X + (long)blockIdx.x * E_;
  float* y = Y + (long)blockIdx.x * E_;
  __shared__ float red[256];
  const int t = threadIdx.x;
  float v = x[t];
  red[t] = v; __syncthreads();
  for (int s = 128; s > 0; s >>= 1) { if (t < s) red[t] += red[t + s]; __syncthreads(); }
  float mu = red[0] / (float)E_; __syncthreads();
  float d = v - mu;
  red[t] = d * d; __syncthreads();
  for (int s = 128; s > 0; s >>= 1) { if (t < s) red[t] += red[t + s]; __syncthreads(); }
  float o = d * rsqrtf(red[0] / (float)E_ + 1e-5f) * w[t] + bb[t];
  y[t] = fmaxf(o, 0.f);
}

// ---------- silu(a)*b, per valid row ----------
__global__ __launch_bounds__(256)
void silu_gate_rows(const float* __restrict__ A, const float* __restrict__ B,
                    float* __restrict__ G, long bstride)
{
  long off = (long)blockIdx.z * bstride + (long)blockIdx.x * INNER;
  for (int c = threadIdx.x; c < INNER; c += 256) {
    float a = A[off + c];
    G[off + c] = (a / (1.f + __expf(-a))) * B[off + c];
  }
}

// ---------- bias (+relu) in place ----------
__global__ void bias_act_rows(float* __restrict__ X, const float* __restrict__ bias,
                              int cols, int act)
{
  float* row = X + (long)blockIdx.x * cols;
  for (int c = threadIdx.x; c < cols; c += blockDim.x) {
    float v = row[c] + bias[c];
    row[c] = act ? fmaxf(v, 0.f) : v;
  }
}

// ---------- connectivity mask (gmask incl. prompt row/col = 1) ----------
__global__ __launch_bounds__(256)
void build_mask(const float* __restrict__ nodes, unsigned char* __restrict__ mask,
                int n, int nm, int mstride, long mbatch)
{
  const int i = blockIdx.x, b = blockIdx.z;
  unsigned char* mrow = mask + (long)b * mbatch + (long)i * mstride;
  bool ip = false; float pvmin = 0.f, pvmax = 0.f;
  if (i < n) {
    const float* n0 = nodes + (long)i * 16;            // types from batch 0
    float t0 = n0[0], t1 = n0[1], t2 = n0[2];
    int ty = (t0 >= t1 && t0 >= t2) ? 0 : (t1 >= t2 ? 1 : 2);
    ip = (ty == 1) || (ty == 0);
    const float* nb = nodes + ((long)b * n + i) * 16;
    pvmin = nb[5]; pvmax = nb[6];
  }
  for (int j = threadIdx.x; j < nm; j += 256) {
    unsigned char v = 1;
    if (i < n && j < n) {
      const float* n0j = nodes + (long)j * 16;
      float u0 = n0j[0], u1 = n0j[1], u2 = n0j[2];
      int tyj = (u0 >= u1 && u0 >= u2) ? 0 : (u1 >= u2 ? 1 : 2);
      bool ic = (tyj == 1) || (tyj == 2);
      const float* nbj = nodes + ((long)b * n + j) * 16;
      bool vc = (pvmin <= nbj[4]) && (pvmax >= nbj[3]);
      v = (ip && ic && vc && (i != j)) ? 1 : 0;
    }
    mrow[j] = v;
  }
}

// ---------- strided row copy ----------
__global__ __launch_bounds__(256)
void copy_rows(const float* __restrict__ src, long sB, int sld,
               float* __restrict__ dst, long dB, int dld, int cols)
{
  const float* s = src + (long)blockIdx.z * sB + (long)blockIdx.x * sld;
  float* d = dst + (long)blockIdx.z * dB + (long)blockIdx.x * dld;
  for (int c = threadIdx.x; c < cols; c += 256) d[c] = s[c];
}

__global__ void zero_f32(float* __restrict__ p, long count)
{
  long i = (long)blockIdx.x * blockDim.x + threadIdx.x;
  if (i < count) p[i] = 0.f;
}

// =====================================================================
// Host side
// =====================================================================
static void launch_gemm(hipStream_t st, const float* A, long sAb, int lda,
                        const float* W, int ldb, float* C, long sCb, int ldc,
                        const float* bias, const float* R, long sRb, int ldr,
                        int M, int N, int K, int act, int nbatch)
{
  dim3 g(cdiv(N, 64), cdiv(M, 64), nbatch);
  wmma_gemm<<<g, 128, 0, st>>>(A, sAb, lda, W, ldb, C, sCb, ldc,
                               bias, R, sRb, ldr, M, N, K, act, 0);
}

struct WsPtrs {
  float *XSP, *XGL, *Q, *K, *V, *CTX, *HPRE, *HN, *FF1, *FF2, *S;
  float *S1, *M1, *CIN, *COUT, *PROMPT;
  unsigned char* MASK;
};

static void enc_layer(hipStream_t st, WsPtrs& w, float* x, int n,
                      const float* wq, const float* wk, const float* wv,
                      const float* wo, const float* bo,
                      const float* n1, const float* n2,
                      const float* l1, const float* l2, const float* l3,
                      const unsigned char* mask, int mstride, long mbatch,
                      bool sparse)
{
  launch_gemm(st, x, XS, E_, wq, E_, w.Q, XS, E_, nullptr, nullptr, 0, 0, n, E_, E_, 0, BATCH);
  launch_gemm(st, x, XS, E_, wk, E_, w.K, XS, E_, nullptr, nullptr, 0, 0, n, E_, E_, 0, BATCH);
  launch_gemm(st, x, XS, E_, wv, E_, w.V, XS, E_, nullptr, nullptr, 0, 0, n, E_, E_, 0, BATCH);

  const int nt = cdiv(n, 64);
  const float scale = 0.17677669529663687f;   // 1/sqrt(32)
  for (int b = 0; b < BATCH; ++b) {           // S buffer sized per-batch (H heads)
    const float* Qb = w.Q + (long)b * XS;
    const float* Kb = w.K + (long)b * XS;
    const float* Vb = w.V + (long)b * XS;
    const unsigned char* mb = mask ? mask + (long)b * mbatch : nullptr;
    wmma_attn_scores<<<dim3(nt, nt, H_), 128, 0, st>>>(Qb, Kb, w.S, mb, n, E_, LDS_S, mstride, scale);
    if (sparse)
      topk_softmax_512<<<dim3(NSP, 1, H_), 256, 0, st>>>(w.S, LDS_S);
    else
      softmax_row<<<dim3(n, 1, H_), 256, 0, st>>>(w.S, n, LDS_S);
    wmma_attn_ctx<<<dim3(1, nt, H_), 128, 0, st>>>(w.S, Vb, w.CTX + (long)b * XS, n, LDS_S, E_);
  }
  launch_gemm(st, w.CTX, XS, E_, wo, E_, w.HPRE, XS, E_, bo, x, XS, E_, n, E_, E_, 0, BATCH);
  rms_rows<<<dim3(n, 1, BATCH), 256, 0, st>>>(w.HPRE, w.HN, n1, XS);
  launch_gemm(st, w.HN, XS, E_, l1, INNER, w.FF1, FFS, INNER, nullptr, nullptr, 0, 0, n, INNER, E_, 0, BATCH);
  launch_gemm(st, w.HN, XS, E_, l2, INNER, w.FF2, FFS, INNER, nullptr, nullptr, 0, 0, n, INNER, E_, 0, BATCH);
  silu_gate_rows<<<dim3(n, 1, BATCH), 256, 0, st>>>(w.FF1, w.FF2, w.FF1, FFS);
  launch_gemm(st, w.FF1, FFS, INNER, l3, E_, w.HPRE, XS, E_, nullptr, w.HN, XS, E_, n, E_, INNER, 0, BATCH);
  rms_rows<<<dim3(n, 1, BATCH), 256, 0, st>>>(w.HPRE, x, n2, XS);
}

extern "C" void kernel_launch(void* const* d_in, const int* in_sizes, int n_in,
                              void* d_out, int out_size, void* d_ws, size_t ws_size,
                              hipStream_t stream)
{
  (void)in_sizes; (void)n_in; (void)out_size;
  const float* nodes  = (const float*)d_in[0];
  const float* scalar = (const float*)d_in[1];
  const float* matrix = (const float*)d_in[2];
  const float* emb_w  = (const float*)d_in[3];
  const float* emb_b  = (const float*)d_in[4];
  auto SPp = [&](int i, int k) { return (const float*)d_in[5  + 10 * i + k]; };
  auto GLp = [&](int i, int k) { return (const float*)d_in[35 + 10 * i + k]; };
  auto SFp = [&](int i, int k) { return (const float*)d_in[65 + 2 * i + k]; };
  auto GFp = [&](int i, int k) { return (const float*)d_in[71 + 2 * i + k]; };
  const float* sw1 = (const float*)d_in[75]; const float* sb1 = (const float*)d_in[76];
  const float* sw2 = (const float*)d_in[77]; const float* sb2 = (const float*)d_in[78];
  const float* mw1 = (const float*)d_in[79]; const float* mb1 = (const float*)d_in[80];
  const float* mw2 = (const float*)d_in[81]; const float* mb2 = (const float*)d_in[82];
  const float* fw  = (const float*)d_in[83]; const float* fb  = (const float*)d_in[84];
  const float* lnw = (const float*)d_in[85]; const float* lnb = (const float*)d_in[86];

  // ---- workspace layout (floats) ----
  float* f = (float*)d_ws;
  WsPtrs w;
  long off = 0;
  w.XSP  = f + off; off += BATCH * XS;
  w.XGL  = f + off; off += BATCH * XS;
  w.Q    = f + off; off += BATCH * XS;
  w.K    = f + off; off += BATCH * XS;
  w.V    = f + off; off += BATCH * XS;
  w.CTX  = f + off; off += BATCH * XS;
  w.HPRE = f + off; off += BATCH * XS;
  w.HN   = f + off; off += BATCH * XS;
  w.FF1  = f + off; off += BATCH * FFS;
  w.FF2  = f + off; off += BATCH * FFS;
  w.S    = f + off; off += H_ * SH;          // per-batch reuse
  w.S1   = f + off; off += BATCH * 128;
  w.M1   = f + off; off += BATCH * E_;
  w.CIN  = f + off; off += BATCH * E_;
  w.COUT = f + off; off += BATCH * E_;
  w.PROMPT = f + off; off += BATCH * E_;
  w.MASK = (unsigned char*)(f + off);
  const long mbatch = (long)NPAD * NPAD;     // mask row stride NPAD, per batch
  size_t need = (size_t)off * 4 + (size_t)BATCH * mbatch;
  if (ws_size < need) return;                // workspace too small; bail safely

  // ---- prompt MLP ----
  zero_f32<<<dim3(8), 256, 0, stream>>>(w.M1, BATCH * E_);   // split-K target
  launch_gemm(stream, scalar, 0, 4, sw1, 128, w.S1, 0, 128, sb1, nullptr, 0, 0, 8, 128, 4, 1, 1);
  launch_gemm(stream, w.S1, 0, 128, sw2, 128, w.CIN, 0, E_, sb2, nullptr, 0, 0, 8, 128, 128, 0, 1);
  {
    const int K = NSP * NSP;                 // 262144
    dim3 g(cdiv(E_, 64), 1, 128);            // split-K: 128 chunks of 2048
    wmma_gemm<<<g, 128, 0, stream>>>(matrix, 0, K, mw1, E_, w.M1, 0, E_,
                                     nullptr, nullptr, 0, 0, 8, E_, K, 0, 2048);
  }
  bias_act_rows<<<dim3(8), 256, 0, stream>>>(w.M1, mb1, E_, 1);
  launch_gemm(stream, w.M1, 0, E_, mw2, 128, w.CIN + 128, 0, E_, mb2, nullptr, 0, 0, 8, 128, E_, 0, 1);
  launch_gemm(stream, w.CIN, 0, E_, fw, E_, w.COUT, 0, E_, fb, nullptr, 0, 0, 8, E_, E_, 0, 1);
  ln_relu_rows<<<dim3(8), 256, 0, stream>>>(w.COUT, w.PROMPT, lnw, lnb);

  // ---- embedding + branch init ----
  launch_gemm(stream, nodes, (long)NSP * 16, 16, emb_w, E_, w.XSP, XS, E_,
              emb_b, nullptr, 0, 0, NSP, E_, 16, 0, BATCH);
  copy_rows<<<dim3(NSP, 1, BATCH), 256, 0, stream>>>(w.XSP, XS, E_, w.XGL, XS, E_, E_);
  copy_rows<<<dim3(1, 1, BATCH), 256, 0, stream>>>(w.PROMPT, E_, 0, w.XGL + (long)NSP * E_, XS, 0, E_);

  // ---- connectivity mask ----
  build_mask<<<dim3(NGL, 1, BATCH), 256, 0, stream>>>(nodes, w.MASK, NSP, NGL, NPAD, mbatch);

  // ---- dual-branch encoder stack ----
  for (int i = 0; i < 3; ++i) {
    enc_layer(stream, w, w.XSP, NSP,
              SPp(i,0), SPp(i,1), SPp(i,2), SPp(i,3), SPp(i,4),
              SPp(i,5), SPp(i,6), SPp(i,7), SPp(i,8), SPp(i,9),
              nullptr, 0, 0, /*sparse=*/true);
    enc_layer(stream, w, w.XGL, NGL,
              GLp(i,0), GLp(i,1), GLp(i,2), GLp(i,3), GLp(i,4),
              GLp(i,5), GLp(i,6), GLp(i,7), GLp(i,8), GLp(i,9),
              w.MASK, NPAD, mbatch, /*sparse=*/false);
    launch_gemm(stream, w.XGL, XS, E_, SFp(i,0), E_, w.XSP, XS, E_,
                SFp(i,1), w.XSP, XS, E_, NSP, E_, E_, 0, BATCH);
    if (i < 2)
      launch_gemm(stream, w.XSP, XS, E_, GFp(i,0), E_, w.XGL, XS, E_,
                  GFp(i,1), w.XGL, XS, E_, NSP, E_, E_, 0, BATCH);
  }

  // ---- pack output [8,512,256] ----
  copy_rows<<<dim3(NSP, 1, BATCH), 256, 0, stream>>>(w.XSP, XS, E_,
                                                     (float*)d_out, (long)NSP * E_, E_, E_);
}
